// Qwen3MoeSparseMoeBlock_54228257079976
// MI455X (gfx1250) — matrix operand order repaired and checked by reading, not compiled, not run on hardware
//
#include <hip/hip_runtime.h>
#include <stdint.h>

#define H_DIM 2048
#define I_DIM 768
#define NEXP  32
#define TOPK  8
#define NTOK  1024
#define CAP   512   // 2 * (NTOK*TOPK / NEXP)

typedef __attribute__((ext_vector_type(16))) __bf16 v16bf;
typedef __attribute__((ext_vector_type(8)))  __bf16 v8bf;
typedef __attribute__((ext_vector_type(8)))  float  v8f;

__device__ __forceinline__ __bf16 bf(float f) { return (__bf16)f; }

__device__ __forceinline__ float warp_max(float v) {
#pragma unroll
  for (int o = 16; o; o >>= 1) v = fmaxf(v, __shfl_xor(v, o, 32));
  return v;
}
__device__ __forceinline__ float warp_sum(float v) {
#pragma unroll
  for (int o = 16; o; o >>= 1) v += __shfl_xor(v, o, 32);
  return v;
}

// gfx1250 async copy: global -> LDS, tracked by ASYNCcnt (CDNA5 ISA)
__device__ __forceinline__ void async_ld_b128(uint32_t lds_addr, const void* gaddr) {
  asm volatile("global_load_async_to_lds_b128 %0, %1, off"
               :: "v"(lds_addr), "v"(gaddr) : "memory");
}
__device__ __forceinline__ void wait_asynccnt0() {
  asm volatile("s_wait_asynccnt 0x0" ::: "memory");
}
// pipelined wait: current stage's 2 async loads retired (in-order), next stage's 2 in flight
__device__ __forceinline__ void wait_asynccnt2() {
  asm volatile("s_wait_asynccnt 0x2" ::: "memory");
}
// generic LDS pointer -> 32-bit LDS offset (flat LDS aperture keeps offset in addr[31:0])
__device__ __forceinline__ uint32_t lds_off(const void* p) {
  return (uint32_t)(uintptr_t)p;
}

// ---------------------------------------------------------------- init + x->bf16 cast
__global__ void moe_init(float* __restrict__ out, int* __restrict__ counts,
                         int* __restrict__ tslot, float* __restrict__ wslot,
                         const float* __restrict__ x, __bf16* __restrict__ xbf) {
  const int idx = blockIdx.x * blockDim.x + threadIdx.x;
  const int stride = gridDim.x * blockDim.x;
  for (int i = idx; i < NTOK * H_DIM; i += stride) out[i] = 0.0f;
  // padded bf16 copy of activations; row NTOK is all zeros (sentinel row)
  for (int i = idx; i < (NTOK + 1) * H_DIM; i += stride)
    xbf[i] = (i < NTOK * H_DIM) ? bf(x[i]) : bf(0.0f);
  if (idx < NEXP) counts[idx] = 0;
  for (int i = idx; i < NEXP * CAP; i += stride) { tslot[i] = NTOK; wslot[i] = 0.0f; }
}

// ---------------------------------------------------------------- router
// one wave per token; lane == expert index
__global__ __launch_bounds__(256) void moe_router(const float* __restrict__ x,
                                                  const float* __restrict__ gate_w,
                                                  int* __restrict__ sel,
                                                  float* __restrict__ wsel) {
  const int lane = threadIdx.x & 31;
  const int wid  = threadIdx.x >> 5;
  const int t = blockIdx.x * 8 + wid;
  if (t >= NTOK) return;
  const float* xr = x + (size_t)t * H_DIM;
  float acc = 0.0f;
  for (int h = 0; h < H_DIM; h += 4) {
    acc += xr[h + 0] * gate_w[(h + 0) * NEXP + lane];
    acc += xr[h + 1] * gate_w[(h + 1) * NEXP + lane];
    acc += xr[h + 2] * gate_w[(h + 2) * NEXP + lane];
    acc += xr[h + 3] * gate_w[(h + 3) * NEXP + lane];
  }
  const float mx = warp_max(acc);
  float p = __expf(acc - mx);
  const float denom = warp_sum(p);
  p = p / denom;

  float wk[TOPK];
  int   ek[TOPK];
  float sw = 0.0f;
#pragma unroll
  for (int k = 0; k < TOPK; ++k) {
    const float m = warp_max(p);
    const unsigned long long b = __ballot(p == m);
    const int src = __ffsll((unsigned long long)b) - 1;
    ek[k] = src;
    wk[k] = m;
    sw += m;
    if (lane == src) p = -1.0f;
  }
  if (lane == 0) {
    const float inv = 1.0f / sw;
#pragma unroll
    for (int k = 0; k < TOPK; ++k) {
      sel[t * TOPK + k]  = ek[k];
      wsel[t * TOPK + k] = wk[k] * inv;
    }
  }
}

// ---------------------------------------------------------------- dispatch
__global__ void moe_dispatch(const int* __restrict__ sel, const float* __restrict__ wsel,
                             int* __restrict__ counts, int* __restrict__ tslot,
                             float* __restrict__ wslot) {
  const int p = blockIdx.x * blockDim.x + threadIdx.x;
  if (p >= NTOK * TOPK) return;
  const int e = sel[p];
  const int pos = atomicAdd(&counts[e], 1);
  if (pos < CAP) {
    tslot[e * CAP + pos] = p / TOPK;
    wslot[e * CAP + pos] = wsel[p];
  }
}

// ---------------------------------------------------------------- fragment helper
__device__ __forceinline__ v16bf frag16(const __bf16* lo_p, const __bf16* hi_p) {
  v8bf lo = *(const v8bf*)lo_p;
  v8bf hi = *(const v8bf*)hi_p;
  return __builtin_shufflevector(lo, hi, 0, 1, 2, 3, 4, 5, 6, 7, 8, 9, 10, 11, 12, 13, 14, 15);
}

// ---------------------------------------------------------------- GEMM1: g,u + SwiGLU
// tile: M=128, N=64, K_TILE=64 (2 wmma k-steps); 8 waves: 4 over M, 2 over N.
// Double-buffered LDS; A tiles stream via async copies one stage ahead; B tiles
// (fp32 weights -> bf16) are register-pipelined one stage ahead.
__global__ __launch_bounds__(256) void moe_gemm1(const __bf16* __restrict__ xbf,
                                                 const float* __restrict__ w_gate_g,
                                                 const float* __restrict__ w_up_g,
                                                 const int* __restrict__ tslot,
                                                 __bf16* __restrict__ hmid) {
  __shared__ __attribute__((aligned(16))) __bf16 As[2][128][72];
  __shared__ __attribute__((aligned(16))) __bf16 Bg[2][64][72];
  __shared__ __attribute__((aligned(16))) __bf16 Bu[2][64][72];

  const int tid = threadIdx.x;
  const int e  = blockIdx.z;
  const int m0 = blockIdx.y * 128;
  const int n0 = blockIdx.x * 64;

  // A staging: async gathered rows of bf16 activations, 2 x b128 chunks per thread
  const int arow = tid >> 1;
  const int acol = (tid & 1) * 32;
  const int atok = tslot[e * CAP + m0 + arow];          // sentinel NTOK -> zero row
  const __bf16* agp = xbf + (size_t)atok * H_DIM + acol;
  uint32_t alds[2][2];
  alds[0][0] = lds_off(&As[0][arow][acol]);
  alds[0][1] = lds_off(&As[0][arow][acol + 16]);
  alds[1][0] = lds_off(&As[1][arow][acol]);
  alds[1][1] = lds_off(&As[1][arow][acol + 16]);

  // B staging: 64 k-rows x 64 n-cols fp32 -> bf16, 16 per thread per matrix
  const int bk = tid >> 2;
  const int bn = (tid & 3) * 16;
  const float* gptr = w_gate_g + ((size_t)e * H_DIM + bk) * I_DIM + n0 + bn;
  const float* uptr = w_up_g   + ((size_t)e * H_DIM + bk) * I_DIM + n0 + bn;

  const int lane = tid & 31;
  const int wid  = tid >> 5;
  const int wm = (wid & 3) * 32;   // 4 waves over M
  const int wn = (wid >> 2) * 32;  // 2 waves over N
  const int r  = lane & 15;
  const int hh = lane >> 4;

  v8f accg[2][2] = {};
  v8f accu[2][2] = {};

  // prologue: stage 0 in flight
  async_ld_b128(alds[0][0], agp);
  async_ld_b128(alds[0][1], agp + 16);
  float rg[16], ru[16];
#pragma unroll
  for (int j = 0; j < 16; ++j) { rg[j] = gptr[j]; ru[j] = uptr[j]; }

  for (int k0 = 0; k0 < H_DIM; k0 += 64) {
    const int cur = (k0 >> 6) & 1;
    const int nxt = cur ^ 1;
    const bool has_next = (k0 + 64) < H_DIM;

    if (has_next) {  // issue next A copy into the other buffer (free since last barrier)
      async_ld_b128(alds[nxt][0], agp + k0 + 64);
      async_ld_b128(alds[nxt][1], agp + k0 + 64 + 16);
    }
    // commit staged B registers to LDS
#pragma unroll
    for (int j = 0; j < 16; ++j) {
      Bg[cur][bn + j][bk] = bf(rg[j]);
      Bu[cur][bn + j][bk] = bf(ru[j]);
    }
    // prefetch next B tiles into registers (overlaps with compute below)
    if (has_next) {
      const float* gp = gptr + (size_t)(k0 + 64) * I_DIM;
      const float* up = uptr + (size_t)(k0 + 64) * I_DIM;
#pragma unroll
      for (int j = 0; j < 16; ++j) { rg[j] = gp[j]; ru[j] = up[j]; }
    }
    if (has_next) wait_asynccnt2(); else wait_asynccnt0();
    __syncthreads();

#pragma unroll
    for (int kk = 0; kk < 64; kk += 32) {
      v16bf a[2], bgf[2], buf[2];
#pragma unroll
      for (int ms = 0; ms < 2; ++ms) {
        const __bf16* ap = &As[cur][wm + ms * 16 + r][kk];
        a[ms] = frag16(ap + 8 * hh, ap + 16 + 8 * hh);
      }
#pragma unroll
      for (int ns = 0; ns < 2; ++ns) {
        const __bf16* g2 = &Bg[cur][wn + ns * 16 + r][kk];
        bgf[ns] = frag16(g2 + 8 * hh, g2 + 16 + 8 * hh);
        const __bf16* u2 = &Bu[cur][wn + ns * 16 + r][kk];
        buf[ns] = frag16(u2 + 8 * hh, u2 + 16 + 8 * hh);
      }
#pragma unroll
      for (int ms = 0; ms < 2; ++ms)
#pragma unroll
        for (int ns = 0; ns < 2; ++ns) {
          accg[ms][ns] = __builtin_amdgcn_wmma_f32_16x16x32_bf16(
              false, a[ms], false, bgf[ns], (short)0, accg[ms][ns], false, false);
          accu[ms][ns] = __builtin_amdgcn_wmma_f32_16x16x32_bf16(
              false, a[ms], false, buf[ns], (short)0, accu[ms][ns], false, false);
        }
    }
    __syncthreads();
  }

  // epilogue: hmid = silu(g) * u, stored bf16 (fast rcp: no IEEE division needed)
#pragma unroll
  for (int ms = 0; ms < 2; ++ms)
#pragma unroll
    for (int j = 0; j < 8; ++j) {
      const int row = m0 + wm + ms * 16 + 8 * hh + j;
#pragma unroll
      for (int ns = 0; ns < 2; ++ns) {
        const int col = n0 + wn + ns * 16 + r;
        const float g = accg[ms][ns][j];
        const float u = accu[ms][ns][j];
        const float s = g * __builtin_amdgcn_rcpf(1.0f + __expf(-g));
        hmid[((size_t)e * CAP + row) * I_DIM + col] = bf(s * u);
      }
    }
}

// ---------------------------------------------------------------- GEMM2: down + combine
__global__ __launch_bounds__(256) void moe_gemm2(const __bf16* __restrict__ hmid,
                                                 const float* __restrict__ w_down_g,
                                                 const int* __restrict__ tslot,
                                                 const float* __restrict__ wslot,
                                                 float* __restrict__ out) {
  __shared__ __attribute__((aligned(16))) __bf16 As[2][128][72];
  __shared__ __attribute__((aligned(16))) __bf16 Bd[2][64][72];

  const int tid = threadIdx.x;
  const int e  = blockIdx.z;
  const int m0 = blockIdx.y * 128;
  const int n0 = blockIdx.x * 64;

  const int arow = tid >> 1;
  const int acol = (tid & 1) * 32;
  const __bf16* agp = hmid + ((size_t)e * CAP + m0 + arow) * I_DIM + acol;
  uint32_t alds[2][2];
  alds[0][0] = lds_off(&As[0][arow][acol]);
  alds[0][1] = lds_off(&As[0][arow][acol + 16]);
  alds[1][0] = lds_off(&As[1][arow][acol]);
  alds[1][1] = lds_off(&As[1][arow][acol + 16]);

  const int bk = tid >> 2;
  const int bn = (tid & 3) * 16;
  const float* dptr = w_down_g + ((size_t)e * I_DIM + bk) * H_DIM + n0 + bn;

  const int lane = tid & 31;
  const int wid  = tid >> 5;
  const int wm = (wid & 3) * 32;
  const int wn = (wid >> 2) * 32;
  const int r  = lane & 15;
  const int hh = lane >> 4;

  v8f acc[2][2] = {};

  // prologue
  async_ld_b128(alds[0][0], agp);
  async_ld_b128(alds[0][1], agp + 16);
  float rd[16];
#pragma unroll
  for (int j = 0; j < 16; ++j) rd[j] = dptr[j];

  for (int k0 = 0; k0 < I_DIM; k0 += 64) {
    const int cur = (k0 >> 6) & 1;
    const int nxt = cur ^ 1;
    const bool has_next = (k0 + 64) < I_DIM;

    if (has_next) {
      async_ld_b128(alds[nxt][0], agp + k0 + 64);
      async_ld_b128(alds[nxt][1], agp + k0 + 64 + 16);
    }
#pragma unroll
    for (int j = 0; j < 16; ++j) Bd[cur][bn + j][bk] = bf(rd[j]);
    if (has_next) {
      const float* dp = dptr + (size_t)(k0 + 64) * H_DIM;
#pragma unroll
      for (int j = 0; j < 16; ++j) rd[j] = dp[j];
    }
    if (has_next) wait_asynccnt2(); else wait_asynccnt0();
    __syncthreads();

#pragma unroll
    for (int kk = 0; kk < 64; kk += 32) {
      v16bf a[2], b[2];
#pragma unroll
      for (int ms = 0; ms < 2; ++ms) {
        const __bf16* ap = &As[cur][wm + ms * 16 + r][kk];
        a[ms] = frag16(ap + 8 * hh, ap + 16 + 8 * hh);
      }
#pragma unroll
      for (int ns = 0; ns < 2; ++ns) {
        const __bf16* bp = &Bd[cur][wn + ns * 16 + r][kk];
        b[ns] = frag16(bp + 8 * hh, bp + 16 + 8 * hh);
      }
#pragma unroll
      for (int ms = 0; ms < 2; ++ms)
#pragma unroll
        for (int ns = 0; ns < 2; ++ns)
          acc[ms][ns] = __builtin_amdgcn_wmma_f32_16x16x32_bf16(
              false, a[ms], false, b[ns], (short)0, acc[ms][ns], false, false);
    }
    __syncthreads();
  }

  // weighted scatter-add back to tokens
#pragma unroll
  for (int ms = 0; ms < 2; ++ms)
#pragma unroll
    for (int j = 0; j < 8; ++j) {
      const int row = m0 + wm + ms * 16 + 8 * hh + j;
      const int tok = tslot[e * CAP + row];
      if (tok >= NTOK) continue;
      const float w = wslot[e * CAP + row];
#pragma unroll
      for (int ns = 0; ns < 2; ++ns) {
        const int col = n0 + wn + ns * 16 + r;
        atomicAdd(&out[(size_t)tok * H_DIM + col], w * acc[ms][ns][j]);
      }
    }
}

// ---------------------------------------------------------------- launch
extern "C" void kernel_launch(void* const* d_in, const int* in_sizes, int n_in,
                              void* d_out, int out_size, void* d_ws, size_t ws_size,
                              hipStream_t stream) {
  const float* x      = (const float*)d_in[0];  // [1,1024,2048]
  const float* gate_w = (const float*)d_in[1];  // [2048,32]
  const float* w_gate = (const float*)d_in[2];  // [32,2048,768]
  const float* w_up   = (const float*)d_in[3];  // [32,2048,768]
  const float* w_down = (const float*)d_in[4];  // [32,768,2048]
  float* out = (float*)d_out;

  char* ws = (char*)d_ws;
  size_t off = 0;
  int*   sel    = (int*)  (ws + off); off += (size_t)NTOK * TOPK * 4;
  float* wsel   = (float*)(ws + off); off += (size_t)NTOK * TOPK * 4;
  int*   counts = (int*)  (ws + off); off += 1024;
  int*   tslot  = (int*)  (ws + off); off += (size_t)NEXP * CAP * 4;
  float* wslot  = (float*)(ws + off); off += (size_t)NEXP * CAP * 4;
  __bf16* xbf   = (__bf16*)(ws + off); off += (size_t)(NTOK + 1) * H_DIM * 2;
  __bf16* hmid  = (__bf16*)(ws + off);  // E*C*I bf16 = 24 MB

  moe_init<<<dim3(1024), dim3(256), 0, stream>>>(out, counts, tslot, wslot, x, xbf);
  moe_router<<<dim3(NTOK / 8), dim3(256), 0, stream>>>(x, gate_w, sel, wsel);
  moe_dispatch<<<dim3(NTOK * TOPK / 256), dim3(256), 0, stream>>>(sel, wsel, counts, tslot, wslot);
  moe_gemm1<<<dim3(I_DIM / 64, CAP / 128, NEXP), dim3(256), 0, stream>>>(xbf, w_gate, w_up, tslot, hmid);
  moe_gemm2<<<dim3(H_DIM / 64, CAP / 128, NEXP), dim3(256), 0, stream>>>(hmid, w_down, tslot, wslot, out);
}
